// MoELayer_16423954940130
// MI455X (gfx1250) — compile-verified, matrix operand order checked by reading, not run on hardware
//
#include <hip/hip_runtime.h>
#include <cstdint>

#define T_TOKENS 4096
#define D_DIM    1024
#define E_EXP    8
#define FF_DIM   4096
#define TOPK     2

#define BM 64       // rows (token-expert entries) per block
#define BN 128      // output columns per block (8 waves x 16)
#define KC 32       // K chunk = WMMA K for bf16
#define LDSTR 40    // padded LDS row stride (bf16 elems); 80B = 5x16B, keeps b128 alignment

typedef __attribute__((ext_vector_type(16))) __bf16 v16bf;
typedef __attribute__((ext_vector_type(8)))  __bf16 v8bf;
typedef __attribute__((ext_vector_type(8)))  float  v8f;

// Low 32 bits of a generic pointer to __shared__ = LDS byte offset (aperture rule).
__device__ __forceinline__ unsigned lds_off(const void* p) {
  return (unsigned)(uintptr_t)p;
}

// CDNA5 async DMA: global -> LDS, 16B per lane, tracked by ASYNCcnt.
__device__ __forceinline__ void async_b128(unsigned lds, const void* g) {
  asm volatile("global_load_async_to_lds_b128 %0, %1, off"
               :: "v"(lds), "v"(g) : "memory");
}
__device__ __forceinline__ void wait_async0() {
  asm volatile("s_wait_asynccnt 0x0" ::: "memory");
}

// Fragment loader for 16-bit A (16x32) and B (32x16 from [N][K] tile).
// Per ISA 7.12.2: lane r=lane%16 selects row; half h=lane/16 selects k-groups:
// VGPRs 0..3 hold k = h*8+0..7, VGPRs 4..7 hold k = 16+h*8+0..7.
__device__ __forceinline__ v16bf load_frag(const __bf16* base, int lr, int lh) {
  union U { v16bf v; struct { v8bf lo; v8bf hi; } s; } u;
  const __bf16* p = base + lr * LDSTR + lh * 8;
  u.s.lo = *(const v8bf*)(p);
  u.s.hi = *(const v8bf*)(p + 16);
  return u.v;
}

// ---------------------------------------------------------------- zero init
__global__ void moe_zero_kernel(float* __restrict__ out, int n) {
  int i = blockIdx.x * blockDim.x + threadIdx.x;
  if (i < n) out[i] = 0.0f;
}

// ---------------------------------------------------------------- router
__global__ void moe_router_kernel(const float* __restrict__ x,
                                  const float* __restrict__ Wr,
                                  int* __restrict__ topi, float* __restrict__ topv,
                                  float* __restrict__ probs) {
  int t = blockIdx.x * blockDim.x + threadIdx.x;
  if (t >= T_TOKENS) return;
  float acc[E_EXP];
#pragma unroll
  for (int e = 0; e < E_EXP; ++e) acc[e] = 0.0f;
  const float* xr = x + (size_t)t * D_DIM;
  for (int d = 0; d < D_DIM; d += 4) {
    float4 xv = *(const float4*)(xr + d);
#pragma unroll
    for (int e = 0; e < E_EXP; ++e) {
      float4 wv = *(const float4*)(Wr + (size_t)e * D_DIM + d);
      acc[e] += xv.x * wv.x + xv.y * wv.y + xv.z * wv.z + xv.w * wv.w;
    }
  }
  float m = acc[0];
#pragma unroll
  for (int e = 1; e < E_EXP; ++e) m = fmaxf(m, acc[e]);
  float p[E_EXP]; float s = 0.0f;
#pragma unroll
  for (int e = 0; e < E_EXP; ++e) { p[e] = __expf(acc[e] - m); s += p[e]; }
  float inv = __builtin_amdgcn_rcpf(s);
#pragma unroll
  for (int e = 0; e < E_EXP; ++e) { p[e] *= inv; probs[(size_t)t * E_EXP + e] = p[e]; }
  int i0 = 0;
#pragma unroll
  for (int e = 1; e < E_EXP; ++e) if (p[e] > p[i0]) i0 = e;
  int i1 = (i0 == 0) ? 1 : 0;
#pragma unroll
  for (int e = 0; e < E_EXP; ++e) if (e != i0 && p[e] > p[i1]) i1 = e;
  float v0 = p[i0], v1 = p[i1];
  float rden = __builtin_amdgcn_rcpf(v0 + v1 + 1e-6f);
  topi[t * 2 + 0] = i0; topi[t * 2 + 1] = i1;
  topv[t * 2 + 0] = v0 * rden; topv[t * 2 + 1] = v1 * rden;
}

// --------------------------------------------- deterministic expert scatter
__global__ void moe_scatter_kernel(const int* __restrict__ topi,
                                   const float* __restrict__ topv,
                                   int* __restrict__ counts, int* __restrict__ offsets,
                                   int* __restrict__ rowToken, float* __restrict__ rowWeight) {
  __shared__ int sc[E_EXP];
  int e = threadIdx.x;
  if (e < E_EXP) {
    int c = 0;
    for (int i = 0; i < T_TOKENS * TOPK; ++i) c += (topi[i] == e) ? 1 : 0;
    sc[e] = c;
  }
  __syncthreads();
  if (e < E_EXP) {
    int off = 0;
    for (int j = 0; j < e; ++j) off += sc[j];
    counts[e] = sc[e];
    offsets[e] = off;
    int pos = off;
    for (int i = 0; i < T_TOKENS * TOPK; ++i) {
      if (topi[i] == e) {
        rowToken[pos] = i >> 1;
        rowWeight[pos] = topv[i];
        pos++;
      }
    }
  }
  if (e == 0) {
    int tot = 0;
    for (int j = 0; j < E_EXP; ++j) tot += sc[j];
    offsets[E_EXP] = tot;
  }
}

// ---------------------------------------------------------------- x -> bf16
__global__ void moe_cvtx_kernel(const float* __restrict__ x, __bf16* __restrict__ xb) {
  int i = (blockIdx.x * blockDim.x + threadIdx.x) * 4;
  float4 v = *(const float4*)(x + i);
  xb[i + 0] = (__bf16)v.x;
  xb[i + 1] = (__bf16)v.y;
  xb[i + 2] = (__bf16)v.z;
  xb[i + 3] = (__bf16)v.w;
}

// ------------------------------- weights: fp32 [E][K][N] -> bf16 [E][N][K]
// Tiled LDS transpose: coalesced reads and writes. One-time pass; afterwards
// the GEMM inner loops stream half the bytes and stage with pure b128 DMA.
__global__ __launch_bounds__(256)
void moe_wtrans_kernel(const float* __restrict__ src, __bf16* __restrict__ dst,
                       int K, int N) {
  __shared__ float tile[32][33];
  int e = blockIdx.z;
  const float* s = src + (size_t)e * K * N;
  __bf16*      d = dst + (size_t)e * K * N;
  int n0 = blockIdx.x * 32, k0 = blockIdx.y * 32;
  int tx = threadIdx.x & 31;     // inner (coalesced) index
  int ty = threadIdx.x >> 5;     // 8 rows per pass
#pragma unroll
  for (int r = 0; r < 32; r += 8)
    tile[ty + r][tx] = s[(size_t)(k0 + ty + r) * N + n0 + tx];   // tile[k][n]
  __syncthreads();
#pragma unroll
  for (int r = 0; r < 32; r += 8)
    d[(size_t)(n0 + ty + r) * K + k0 + tx] = (__bf16)tile[tx][ty + r];
}

// ------------------------------------------------- phase A: fused fc1 + gate
// act[row, ff] = silu(x@Wg + bg) * (x@W1 + b1), rows grouped by expert (bf16)
__global__ __launch_bounds__(256)
void moe_fc1_gate_kernel(const __bf16* __restrict__ xb,
                         const __bf16* __restrict__ W1T, const float* __restrict__ b1,
                         const __bf16* __restrict__ WgT, const float* __restrict__ bg,
                         const int* __restrict__ offsets, const int* __restrict__ counts,
                         const int* __restrict__ rowToken,
                         __bf16* __restrict__ act) {
  int e = blockIdx.z;
  int n0 = blockIdx.y * BN;
  int cnt = counts[e];
  int r0 = blockIdx.x * BM;
  if (r0 >= cnt) return;
  int seg = offsets[e];

  __shared__ __bf16 As[2][BM * LDSTR];
  __shared__ __bf16 W1s[2][BN * LDSTR];
  __shared__ __bf16 Wgs[2][BN * LDSTR];

  int tid  = threadIdx.x;
  int wave = tid >> 5, lane = tid & 31;
  int lr = lane & 15, lh = lane >> 4;
  int nW = wave * 16;

  // A staging: 64 rows x 32 bf16 = 4KB; one 16B DMA per thread (gathered rows).
  int arow = tid >> 2, achk = tid & 3;
  int grow = r0 + arow;
  int tok  = rowToken[seg + (grow < cnt ? grow : cnt - 1)];
  const __bf16* aG = xb + (size_t)tok * D_DIM + achk * 8;
  unsigned aL = lds_off(&As[0][arow * LDSTR + achk * 8]);

  // W staging: 128 rows x 32 bf16 = 8KB per matrix; two 16B DMAs per thread.
  int wrow = tid >> 1, wc = tid & 1;  // wc selects 32B half of the 64B row
  const __bf16* w1G = W1T + (size_t)e * D_DIM * FF_DIM + (size_t)(n0 + wrow) * D_DIM + wc * 16;
  const __bf16* wgG = WgT + (size_t)e * D_DIM * FF_DIM + (size_t)(n0 + wrow) * D_DIM + wc * 16;
  unsigned w1L = lds_off(&W1s[0][wrow * LDSTR + wc * 16]);
  unsigned wgL = lds_off(&Wgs[0][wrow * LDSTR + wc * 16]);

  const unsigned ABUF = sizeof(As[0]), WBUF = sizeof(W1s[0]);

  auto stage = [&](int buf, int k0) {
    async_b128(aL  + buf * ABUF,      aG  + k0);
    async_b128(w1L + buf * WBUF,      w1G + k0);
    async_b128(w1L + buf * WBUF + 16, w1G + k0 + 8);
    async_b128(wgL + buf * WBUF,      wgG + k0);
    async_b128(wgL + buf * WBUF + 16, wgG + k0 + 8);
  };

  v8f acc_h[4] = {}, acc_g[4] = {};

  stage(0, 0);
  int buf = 0;
  for (int k0 = 0; k0 < D_DIM; k0 += KC, buf ^= 1) {
    wait_async0();            // this buffer's DMA complete (per wave)
    __syncthreads();          // visible to all waves; prior reads of other buf done
    if (k0 + KC < D_DIM) stage(buf ^ 1, k0 + KC);   // overlap next DMA with compute

    v16bf bf1 = load_frag(&W1s[buf][nW * LDSTR], lr, lh);
    v16bf bfg = load_frag(&Wgs[buf][nW * LDSTR], lr, lh);
#pragma unroll
    for (int mi = 0; mi < 4; ++mi) {
      v16bf af = load_frag(&As[buf][mi * 16 * LDSTR], lr, lh);
      acc_h[mi] = __builtin_amdgcn_wmma_f32_16x16x32_bf16(false, af, false, bf1,
                                                          (short)0, acc_h[mi], false, false);
      acc_g[mi] = __builtin_amdgcn_wmma_f32_16x16x32_bf16(false, af, false, bfg,
                                                          (short)0, acc_g[mi], false, false);
    }
  }

  // epilogue: bias + silu(g)*h -> bf16 act (fast rcp: one v_rcp_f32)
  int nGlob = n0 + nW + lr;
  float b1v = b1[(size_t)e * FF_DIM + nGlob];
  float bgv = bg[(size_t)e * FF_DIM + nGlob];
#pragma unroll
  for (int mi = 0; mi < 4; ++mi) {
#pragma unroll
    for (int v = 0; v < 8; ++v) {
      int m = mi * 16 + lh * 8 + v;     // C layout: VGPR v, half lh -> row
      int gr = r0 + m;
      float h = acc_h[mi][v] + b1v;
      float g = acc_g[mi][v] + bgv;
      float a = g * __builtin_amdgcn_rcpf(1.0f + __expf(-g)) * h;
      if (gr < cnt)
        act[(size_t)(seg + gr) * FF_DIM + nGlob] = (__bf16)a;
    }
  }
}

// ------------------------------------------------- phase B: fc2 + combine
// out[token, :] += w * (act @ W2[e] + b2[e])
__global__ __launch_bounds__(256)
void moe_fc2_kernel(const __bf16* __restrict__ act,
                    const __bf16* __restrict__ W2T, const float* __restrict__ b2,
                    const int* __restrict__ offsets, const int* __restrict__ counts,
                    const int* __restrict__ rowToken, const float* __restrict__ rowWeight,
                    float* __restrict__ out) {
  int e = blockIdx.z;
  int n0 = blockIdx.y * BN;
  int cnt = counts[e];
  int r0 = blockIdx.x * BM;
  if (r0 >= cnt) return;
  int seg = offsets[e];

  __shared__ __bf16 As[2][BM * LDSTR];
  __shared__ __bf16 Ws[2][BN * LDSTR];
  __shared__ int   rtok[BM];
  __shared__ float rwgt[BM];

  int tid  = threadIdx.x;
  int wave = tid >> 5, lane = tid & 31;
  int lr = lane & 15, lh = lane >> 4;
  int nW = wave * 16;

  int arow = tid >> 2, achk = tid & 3;
  int grow = r0 + arow;
  const __bf16* aG = act + (size_t)(seg + (grow < cnt ? grow : cnt - 1)) * FF_DIM + achk * 8;
  unsigned aL = lds_off(&As[0][arow * LDSTR + achk * 8]);

  int wrow = tid >> 1, wc = tid & 1;
  const __bf16* wG = W2T + (size_t)e * FF_DIM * D_DIM + (size_t)(n0 + wrow) * FF_DIM + wc * 16;
  unsigned wL = lds_off(&Ws[0][wrow * LDSTR + wc * 16]);

  const unsigned ABUF = sizeof(As[0]), WBUF = sizeof(Ws[0]);

  if (tid < BM) {
    int gr = r0 + tid;
    if (gr < cnt) { rtok[tid] = rowToken[seg + gr]; rwgt[tid] = rowWeight[seg + gr]; }
    else          { rtok[tid] = 0;                  rwgt[tid] = 0.0f; }
  }

  auto stage = [&](int buf, int k0) {
    async_b128(aL + buf * ABUF,      aG + k0);
    async_b128(wL + buf * WBUF,      wG + k0);
    async_b128(wL + buf * WBUF + 16, wG + k0 + 8);
  };

  v8f acc[4] = {};

  stage(0, 0);
  int buf = 0;
  for (int k0 = 0; k0 < FF_DIM; k0 += KC, buf ^= 1) {
    wait_async0();
    __syncthreads();
    if (k0 + KC < FF_DIM) stage(buf ^ 1, k0 + KC);

    v16bf bf = load_frag(&Ws[buf][nW * LDSTR], lr, lh);
#pragma unroll
    for (int mi = 0; mi < 4; ++mi) {
      v16bf af = load_frag(&As[buf][mi * 16 * LDSTR], lr, lh);
      acc[mi] = __builtin_amdgcn_wmma_f32_16x16x32_bf16(false, af, false, bf,
                                                        (short)0, acc[mi], false, false);
    }
  }

  int nGlob = n0 + nW + lr;
  float b2v = b2[(size_t)e * D_DIM + nGlob];
#pragma unroll
  for (int mi = 0; mi < 4; ++mi) {
#pragma unroll
    for (int v = 0; v < 8; ++v) {
      int m = mi * 16 + lh * 8 + v;
      int gr = r0 + m;
      if (gr < cnt) {
        float val = rwgt[m] * (acc[mi][v] + b2v);
        atomicAdd(&out[(size_t)rtok[m] * D_DIM + nGlob], val);
      }
    }
  }
}

// ------------------------------------------------- aux-loss reductions
__global__ void moe_probsum_kernel(const float* __restrict__ probs, float* __restrict__ probsum) {
  __shared__ float sdata[256];
  int e = blockIdx.x;
  float s = 0.0f;
  for (int t = threadIdx.x; t < T_TOKENS; t += 256) s += probs[(size_t)t * E_EXP + e];
  sdata[threadIdx.x] = s;
  __syncthreads();
  for (int st = 128; st > 0; st >>= 1) {
    if (threadIdx.x < st) sdata[threadIdx.x] += sdata[threadIdx.x + st];
    __syncthreads();
  }
  if (threadIdx.x == 0) probsum[e] = sdata[0];
}

__global__ void moe_lb_kernel(const int* __restrict__ counts,
                              const float* __restrict__ probsum,
                              float* __restrict__ lb_out) {
  if (threadIdx.x == 0 && blockIdx.x == 0) {
    float tot = 0.0f;
    for (int e = 0; e < E_EXP; ++e) tot += (float)counts[e];
    float lb = 0.0f;
    for (int e = 0; e < E_EXP; ++e)
      lb += ((float)counts[e] / (tot + 1e-6f)) * probsum[e];
    lb_out[0] = lb * (float)E_EXP;
  }
}

// ---------------------------------------------------------------- launcher
extern "C" void kernel_launch(void* const* d_in, const int* in_sizes, int n_in,
                              void* d_out, int out_size, void* d_ws, size_t ws_size,
                              hipStream_t stream) {
  const float* x  = (const float*)d_in[0];
  const float* Wr = (const float*)d_in[1];
  const float* W1 = (const float*)d_in[2];
  const float* b1 = (const float*)d_in[3];
  const float* Wg = (const float*)d_in[4];
  const float* bg = (const float*)d_in[5];
  const float* W2 = (const float*)d_in[6];
  const float* b2 = (const float*)d_in[7];
  float* out = (float*)d_out;           // [T*D] out_final, then [1] lb

  char* ws = (char*)d_ws;
  size_t o = 0;
  auto alloc = [&](size_t bytes) { size_t r = o; o += (bytes + 255) & ~(size_t)255; return r; };
  int*    topi      = (int*)   (ws + alloc(sizeof(int)   * T_TOKENS * TOPK));
  float*  topv      = (float*) (ws + alloc(sizeof(float) * T_TOKENS * TOPK));
  float*  probs     = (float*) (ws + alloc(sizeof(float) * T_TOKENS * E_EXP));
  int*    counts    = (int*)   (ws + alloc(sizeof(int)   * E_EXP));
  int*    offsets   = (int*)   (ws + alloc(sizeof(int)   * (E_EXP + 1)));
  int*    rowToken  = (int*)   (ws + alloc(sizeof(int)   * T_TOKENS * TOPK));
  float*  rowWeight = (float*) (ws + alloc(sizeof(float) * T_TOKENS * TOPK));
  float*  probsum   = (float*) (ws + alloc(sizeof(float) * E_EXP));
  __bf16* xb  = (__bf16*)(ws + alloc(sizeof(__bf16) * (size_t)T_TOKENS * D_DIM));
  __bf16* act = (__bf16*)(ws + alloc(sizeof(__bf16) * (size_t)T_TOKENS * TOPK * FF_DIM));
  __bf16* W1T = (__bf16*)(ws + alloc(sizeof(__bf16) * (size_t)E_EXP * D_DIM * FF_DIM));
  __bf16* WgT = (__bf16*)(ws + alloc(sizeof(__bf16) * (size_t)E_EXP * D_DIM * FF_DIM));
  __bf16* W2T = (__bf16*)(ws + alloc(sizeof(__bf16) * (size_t)E_EXP * FF_DIM * D_DIM));
  (void)ws_size; (void)n_in; (void)in_sizes; (void)out_size;

  const int TD = T_TOKENS * D_DIM;
  moe_zero_kernel<<<TD / 256, 256, 0, stream>>>(out, TD);
  moe_router_kernel<<<T_TOKENS / 256, 256, 0, stream>>>(x, Wr, topi, topv, probs);
  moe_scatter_kernel<<<1, 32, 0, stream>>>(topi, topv, counts, offsets, rowToken, rowWeight);
  moe_cvtx_kernel<<<TD / (256 * 4), 256, 0, stream>>>(x, xb);

  // one-time weight bf16 transposes: [E][K][N] fp32 -> [E][N][K] bf16
  moe_wtrans_kernel<<<dim3(FF_DIM / 32, D_DIM / 32, E_EXP), 256, 0, stream>>>(W1, W1T, D_DIM, FF_DIM);
  moe_wtrans_kernel<<<dim3(FF_DIM / 32, D_DIM / 32, E_EXP), 256, 0, stream>>>(Wg, WgT, D_DIM, FF_DIM);
  moe_wtrans_kernel<<<dim3(D_DIM / 32, FF_DIM / 32, E_EXP), 256, 0, stream>>>(W2, W2T, FF_DIM, D_DIM);

  dim3 gA(T_TOKENS / BM, FF_DIM / BN, E_EXP);   // (64, 32, 8); early-exit past counts
  moe_fc1_gate_kernel<<<gA, 256, 0, stream>>>(xb, W1T, b1, WgT, bg,
                                              offsets, counts, rowToken, act);
  dim3 gB(T_TOKENS / BM, D_DIM / BN, E_EXP);    // (64, 8, 8)
  moe_fc2_kernel<<<gB, 256, 0, stream>>>(act, W2T, b2, offsets, counts,
                                         rowToken, rowWeight, out);

  moe_probsum_kernel<<<E_EXP, 256, 0, stream>>>(probs, probsum);
  moe_lb_kernel<<<1, 32, 0, stream>>>(counts, probsum, out + TD);
}
